// DeformableConvolution_5239860101855
// MI455X (gfx1250) — matrix operand order repaired and checked by reading, not compiled, not run on hardware
//
#include <hip/hip_runtime.h>
#include <hip/hip_bf16.h>

// ---------------------------------------------------------------------------
// Deformable 3D convolution for MI455X (gfx1250, wave32, WMMA).
//   B=1, T=4, C=64, H=128, W=128, 3x3x3 taps (27), pad=1.
// Both channel contractions run on v_wmma_f32_16x16x32_bf16.
// ---------------------------------------------------------------------------

typedef __attribute__((ext_vector_type(16))) __bf16 v16bf;
typedef __attribute__((ext_vector_type(8)))  __bf16 v8bf;
typedef __attribute__((ext_vector_type(8)))  float  v8f;

#define DIM_T 4
#define DIM_C 64
#define DIM_H 128
#define DIM_W 128
#define NPOS  (DIM_T * DIM_H * DIM_W)     // 65536
#define NTAPS 27
#define NOFFC 81                          // 3*27 offset channels
#define NOFF_NT 6                         // 81 -> 96 padded N tiles
#define COUT_NT 4                         // 64 outputs -> 4 N tiles

// ---- workspace layout (bytes) ----
#define XC_OFF   ((size_t)0)                                   // bf16 x, (T,H,W,C): 8 MiB
#define XC_BYTES ((size_t)NPOS * DIM_C * 2)
#define OFF_OFF  (XC_OFF + XC_BYTES)                           // f32 offsets, (81, NPOS): ~20 MiB
#define OFF_BYTES ((size_t)NOFFC * NPOS * 4)
#define POW_OFF  (OFF_OFF + OFF_BYTES)                         // packed offset weights
#define POW_BYTES ((size_t)NTAPS * 2 * NOFF_NT * 512 * 2)
#define PW_OFF   (POW_OFF + POW_BYTES)                         // packed main weights
#define PW_BYTES ((size_t)NTAPS * 2 * COUT_NT * 512 * 2)

// ---------------------------------------------------------------------------
// Prep 1: x (B,T,C,H,W) f32 -> xc (T,H,W,C) bf16 (channels-last).
// ---------------------------------------------------------------------------
__global__ void k_cvt_x(const float* __restrict__ x, __bf16* __restrict__ xc) {
    int id = blockIdx.x * blockDim.x + threadIdx.x;      // NPOS*64 threads
    int c   = id & 63;
    int pos = id >> 6;
    int w = pos & 127, h = (pos >> 7) & 127, t = pos >> 14;
    float v = x[(((size_t)t * DIM_C + c) * DIM_H + h) * DIM_W + w];
    xc[id] = (__bf16)v;
}

// ---------------------------------------------------------------------------
// Prep 2/3: pack weights into the exact per-lane 32x16 bf16 B-fragment layout.
// Block = (tap, kb32, ntile); within a block: lane-major, 16 bf16 per lane.
// Lane l holds N = ntile*16 + (l&15), K = kb32*32 + (i>>3)*16 + (l>>4)*8 + (i&7)  (i=0..15).
// ---------------------------------------------------------------------------
__global__ void k_pack_offw(const float* __restrict__ offw, __bf16* __restrict__ dst) {
    int id = blockIdx.x * blockDim.x + threadIdx.x;      // 27*2*6*512 threads
    int i    = id & 15;
    int lane = (id >> 4) & 31;
    int blk  = id >> 9;
    int nt   = blk % NOFF_NT;
    int kb   = (blk / NOFF_NT) & 1;
    int tap  = blk / (NOFF_NT * 2);
    int c    = kb * 32 + (i >> 3) * 16 + (lane >> 4) * 8 + (i & 7); // K = input channel
    int n    = nt * 16 + (lane & 15);                    // N = offset channel
    float v  = (n < NOFFC) ? offw[((size_t)n * DIM_C + c) * NTAPS + tap] : 0.0f;
    dst[id] = (__bf16)v;
}

__global__ void k_pack_w(const float* __restrict__ wgt, __bf16* __restrict__ dst) {
    int id = blockIdx.x * blockDim.x + threadIdx.x;      // 27*2*4*512 threads
    int i    = id & 15;
    int lane = (id >> 4) & 31;
    int blk  = id >> 9;
    int nt   = blk % COUT_NT;
    int kb   = (blk / COUT_NT) & 1;
    int tap  = blk / (COUT_NT * 2);
    int c    = kb * 32 + (i >> 3) * 16 + (lane >> 4) * 8 + (i & 7);
    int o    = nt * 16 + (lane & 15);
    float v  = wgt[((size_t)o * DIM_C + c) * NTAPS + tap];
    dst[id] = (__bf16)v;
}

// Build a 16-element bf16 A fragment from two contiguous 8-channel runs.
__device__ __forceinline__ v16bf make_frag(v8bf lo, v8bf hi) {
    return __builtin_shufflevector(lo, hi, 0,1,2,3,4,5,6,7,8,9,10,11,12,13,14,15);
}

// ---------------------------------------------------------------------------
// Stage 1: offset conv as bf16 WMMA GEMM.
// One wave per 16-position tile. A rows = positions, K = (tap, channel).
// A-fragment per lane: M = lane&15, kgroup = lane>>4; K runs of 8 channels
// at kb*32 + kg*8 and kb*32 + 16 + kg*8 -> two b128 global loads from xc.
// ---------------------------------------------------------------------------
__global__ void __launch_bounds__(128) k_offconv(
        const __bf16* __restrict__ xc, const __bf16* __restrict__ packB,
        const float* __restrict__ offb, float* __restrict__ off_ws) {
    int lane = threadIdx.x & 31;
    int tile = blockIdx.x * 4 + (threadIdx.x >> 5);      // 4096 tiles
    int pos_base = tile * 16;
    int t  = pos_base >> 14;
    int h  = (pos_base >> 7) & 127;
    int wb = pos_base & 127;                              // tile never crosses a row
    int p  = lane & 15;                                   // A-row / position in tile
    int kg = lane >> 4;                                   // K sub-block

    v8f acc[NOFF_NT] = {};
    #pragma unroll
    for (int tap = 0; tap < NTAPS; ++tap) {
        int kt = tap / 9, kh = (tap / 3) % 3, kw = tap % 3;
        int ts = t + kt - 1, hs = h + kh - 1, ws2 = wb + p + kw - 1;
        bool valid = (ts >= 0) & (ts < DIM_T) & (hs >= 0) & (hs < DIM_H) &
                     (ws2 >= 0) & (ws2 < DIM_W);
        const __bf16* src = xc + ((size_t)((ts * DIM_H + hs) * DIM_W + ws2) << 6);
        #pragma unroll
        for (int kb = 0; kb < 2; ++kb) {
            v16bf a;
            if (valid) {
                v8bf lo = *(const v8bf*)(src + kb * 32 + kg * 8);
                v8bf hi = *(const v8bf*)(src + kb * 32 + 16 + kg * 8);
                a = make_frag(lo, hi);
            } else {
                a = v16bf{};                               // zero padding for row M
            }
            const __bf16* bp = packB + (size_t)((tap * 2 + kb) * NOFF_NT) * 512;
            #pragma unroll
            for (int nt = 0; nt < NOFF_NT; ++nt) {
                const v8bf* q = (const v8bf*)(bp + nt * 512 + lane * 16);
                v16bf b = make_frag(q[0], q[1]);
                acc[nt] = __builtin_amdgcn_wmma_f32_16x16x32_bf16(
                    false, a, false, b, (short)0, acc[nt], false, false);
            }
        }
    }
    // C/D layout: N = lane&15, M = r + (lane>>4)*8.
    #pragma unroll
    for (int nt = 0; nt < NOFF_NT; ++nt) {
        int n = nt * 16 + (lane & 15);
        if (n < NOFFC) {
            float bias = offb[n];
            #pragma unroll
            for (int r = 0; r < 8; ++r) {
                int pos = pos_base + r + kg * 8;
                off_ws[(size_t)n * NPOS + pos] = acc[nt][r] + bias;
            }
        }
    }
}

// ---------------------------------------------------------------------------
// Stage 2: deformable sampling (trilinear, zero-fill OOB) + channel
// contraction on WMMA + gamma scale + residual add.
// Lane owns position p = lane&15 and 32 channels in 4 runs of 8:
// run rr -> channels [rr*16 + kg*8, +8). Accumulated in f32 over 8 corners,
// then repacked into two bf16 A fragments (kb=0: runs 0,1; kb=1: runs 2,3).
// ---------------------------------------------------------------------------
__global__ void __launch_bounds__(128) k_deform(
        const __bf16* __restrict__ xc, const __bf16* __restrict__ packW,
        const float* __restrict__ off_ws, const float* __restrict__ x_in,
        const float* __restrict__ gamma, float* __restrict__ out) {
    int lane = threadIdx.x & 31;
    int tile = blockIdx.x * 4 + (threadIdx.x >> 5);
    int pos_base = tile * 16;
    int p  = lane & 15;
    int kg = lane >> 4;
    int pos = pos_base + p;
    int t = pos >> 14, h = (pos >> 7) & 127, w = pos & 127;

    v8f acc[COUT_NT] = {};
    #pragma unroll 1
    for (int tap = 0; tap < NTAPS; ++tap) {
        int kt = tap / 9, kh = (tap / 3) % 3, kw = tap % 3;
        float dt = off_ws[(size_t)(0 * NTAPS + tap) * NPOS + pos];
        float dh = off_ws[(size_t)(1 * NTAPS + tap) * NPOS + pos];
        float dw = off_ws[(size_t)(2 * NTAPS + tap) * NPOS + pos];
        float tf = (float)(t + kt - 1) + dt;
        float hf = (float)(h + kh - 1) + dh;
        float wf = (float)(w + kw - 1) + dw;
        float t0f = floorf(tf), h0f = floorf(hf), w0f = floorf(wf);
        float lt = tf - t0f, lh = hf - h0f, lw = wf - w0f;
        int t0 = (int)t0f, h0 = (int)h0f, w0 = (int)w0f;

        float s[32];
        #pragma unroll
        for (int j = 0; j < 32; ++j) s[j] = 0.0f;

        #pragma unroll
        for (int cz = 0; cz < 2; ++cz) {
            int ti = t0 + cz;
            float wt = cz ? lt : 1.0f - lt;
            #pragma unroll
            for (int cy = 0; cy < 2; ++cy) {
                int hi = h0 + cy;
                float wh = cy ? lh : 1.0f - lh;
                #pragma unroll
                for (int cx = 0; cx < 2; ++cx) {
                    int wi = w0 + cx;
                    float ww = cx ? lw : 1.0f - lw;
                    bool valid = (ti >= 0) & (ti < DIM_T) & (hi >= 0) &
                                 (hi < DIM_H) & (wi >= 0) & (wi < DIM_W);
                    float cw = valid ? (wt * wh * ww) : 0.0f;
                    int tic = ti < 0 ? 0 : (ti > DIM_T - 1 ? DIM_T - 1 : ti);
                    int hic = hi < 0 ? 0 : (hi > DIM_H - 1 ? DIM_H - 1 : hi);
                    int wic = wi < 0 ? 0 : (wi > DIM_W - 1 ? DIM_W - 1 : wi);
                    const __bf16* src =
                        xc + ((size_t)((tic * DIM_H + hic) * DIM_W + wic) << 6);
                    #pragma unroll
                    for (int rr = 0; rr < 4; ++rr) {
                        v8bf v = *(const v8bf*)(src + rr * 16 + kg * 8);
                        #pragma unroll
                        for (int j = 0; j < 8; ++j)
                            s[rr * 8 + j] += cw * (float)v[j];
                    }
                }
            }
        }
        #pragma unroll
        for (int kb = 0; kb < 2; ++kb) {
            v16bf a;
            #pragma unroll
            for (int i = 0; i < 16; ++i)
                a[i] = (__bf16)s[(kb * 2 + (i >> 3)) * 8 + (i & 7)];
            const __bf16* bp = packW + (size_t)((tap * 2 + kb) * COUT_NT) * 512;
            #pragma unroll
            for (int nt = 0; nt < COUT_NT; ++nt) {
                const v8bf* q = (const v8bf*)(bp + nt * 512 + lane * 16);
                v16bf b = make_frag(q[0], q[1]);
                acc[nt] = __builtin_amdgcn_wmma_f32_16x16x32_bf16(
                    false, a, false, b, (short)0, acc[nt], false, false);
            }
        }
    }
    // Epilogue: out(B,T,C,H,W) = gamma * conv + x.
    float g = gamma[0];
    int tb = pos_base >> 14, hb = (pos_base >> 7) & 127, wbb = pos_base & 127;
    #pragma unroll
    for (int nt = 0; nt < COUT_NT; ++nt) {
        int o = nt * 16 + (lane & 15);
        #pragma unroll
        for (int r = 0; r < 8; ++r) {
            int wr = wbb + r + kg * 8;
            size_t idx = (((size_t)tb * DIM_C + o) * DIM_H + hb) * DIM_W + wr;
            out[idx] = g * acc[nt][r] + x_in[idx];
        }
    }
}

// ---------------------------------------------------------------------------
extern "C" void kernel_launch(void* const* d_in, const int* in_sizes, int n_in,
                              void* d_out, int out_size, void* d_ws, size_t ws_size,
                              hipStream_t stream) {
    const float* x     = (const float*)d_in[0];
    const float* offw  = (const float*)d_in[1];
    const float* offb  = (const float*)d_in[2];
    const float* wgt   = (const float*)d_in[3];
    const float* gamma = (const float*)d_in[4];
    float* out = (float*)d_out;

    char* ws = (char*)d_ws;
    __bf16* xc    = (__bf16*)(ws + XC_OFF);
    float*  offws = (float*)(ws + OFF_OFF);
    __bf16* packO = (__bf16*)(ws + POW_OFF);
    __bf16* packW = (__bf16*)(ws + PW_OFF);

    // Prep: channels-last bf16 x, WMMA-layout packed weights.
    k_cvt_x<<<(NPOS * DIM_C) / 256, 256, 0, stream>>>(x, xc);
    k_pack_offw<<<(NTAPS * 2 * NOFF_NT * 512) / 256, 256, 0, stream>>>(offw, packO);
    k_pack_w<<<(NTAPS * 2 * COUT_NT * 512) / 256, 256, 0, stream>>>(wgt, packW);

    // Stage 1: offset conv (GEMM, K = 27 taps x 64 ch).
    k_offconv<<<(NPOS / 16) / 4, 128, 0, stream>>>(xc, packO, offb, offws);

    // Stage 2: deformable sample + output GEMM + residual.
    k_deform<<<(NPOS / 16) / 4, 128, 0, stream>>>(xc, packW, offws, x, gamma, out);
}
